// STATELOSS_8254927143179
// MI455X (gfx1250) — matrix operand order repaired and checked by reading, not compiled, not run on hardware
//
#include <hip/hip_runtime.h>
#include <hip/hip_bf16.h>

// ---------------- types for WMMA ----------------
typedef __attribute__((ext_vector_type(16))) __bf16          v16bf;
typedef __attribute__((ext_vector_type(16))) unsigned short  v16us;
typedef __attribute__((ext_vector_type(8)))  unsigned short  v8us;
typedef __attribute__((ext_vector_type(8)))  float           v8f;

// ---------------- problem constants ----------------
#define BB        2048      // batch
#define DD        2048      // feature dim (K)
#define NW        1000      // NUM_WNIDS
#define GG        372       // groups
#define RMAXC     64
#define EFF_LD    1024      // padded leading dim of eff
#define NROWS_B   1024      // rows of global_states staged (>= all cols computed)
#define NEGF      (-1e30f)

__device__ __forceinline__ unsigned short f2bf(float f) {
    unsigned int u = __float_as_uint(f);
    unsigned int r = u + 0x7FFFu + ((u >> 16) & 1u);   // round-to-nearest-even
    return (unsigned short)(r >> 16);
}

__device__ __forceinline__ v16bf load_frag(const unsigned short* p0,
                                           const unsigned short* p1) {
    v8us a = *(const v8us*)p0;
    v8us b = *(const v8us*)p1;
    v16us u = __builtin_shufflevector(a, b,
        0, 1, 2, 3, 4, 5, 6, 7, 8, 9, 10, 11, 12, 13, 14, 15);
    return __builtin_bit_cast(v16bf, u);
}

// ---------------- fp32 -> bf16 staging ----------------
__global__ void cvt_bf16_kernel(const float* __restrict__ src,
                                unsigned short* __restrict__ dst, int n) {
    int i = blockIdx.x * blockDim.x + threadIdx.x;
    if (i < n) dst[i] = f2bf(src[i]);
}

// ---------------- zero accumulators ----------------
__global__ void init_kernel(float* __restrict__ p, int n) {
    int i = blockIdx.x * blockDim.x + threadIdx.x;
    if (i < n) p[i] = 0.0f;
}

// ---------------- WMMA GEMM: eff = A(2048x2048) * gs[:1024].T + 1 ----------------
// Each wave computes a 32x64 strip: 2 M-tiles x 4 N-tiles = 8 WMMA accumulators.
// Per k-step: 12x16B loads feed 8 v_wmma_f32_16x16x32_bf16 (1.5 loads/WMMA).
// grid: 128 blocks x 256 threads (8 waves) -> 1024 strips = 64 Mstrips * 16 Nstrips.
__global__ void gemm_wmma_kernel(const unsigned short* __restrict__ Abf,
                                 const unsigned short* __restrict__ Bbf,
                                 float* __restrict__ eff) {
    const int lane  = threadIdx.x & 31;
    const int wave  = threadIdx.x >> 5;
    const int strip = blockIdx.x * 8 + wave;       // 0..1023
    const int sm    = strip >> 4;                  // 0..63  -> rows sm*32
    const int sn    = strip & 15;                  // 0..15  -> cols sn*64

    const int mrow = lane & 15;
    const int hi   = lane >> 4;                    // 0 or 1

    // A rows for the two M-tiles
    const unsigned short* arow0 = Abf + (size_t)(sm * 32 + mrow) * DD;
    const unsigned short* arow1 = arow0 + (size_t)16 * DD;
    // B rows (= output columns) for the four N-tiles
    const unsigned short* brow0 = Bbf + (size_t)(sn * 64 + mrow) * DD;
    const unsigned short* brow1 = brow0 + (size_t)16 * DD;
    const unsigned short* brow2 = brow0 + (size_t)32 * DD;
    const unsigned short* brow3 = brow0 + (size_t)48 * DD;

    v8f acc00 = {}, acc01 = {}, acc02 = {}, acc03 = {};
    v8f acc10 = {}, acc11 = {}, acc12 = {}, acc13 = {};

    const int aoff = hi * 16;   // A: lanes 0-15 K=0..15, lanes 16-31 K=16..31 (contiguous)
    const int boff = hi * 16;   // B: lanes 0-15 K=0..15, lanes 16-31 K=16..31 (contiguous)

    for (int k0 = 0; k0 < DD; k0 += 32) {
        v16bf a0 = load_frag(arow0 + k0 + aoff, arow0 + k0 + aoff + 8);
        v16bf a1 = load_frag(arow1 + k0 + aoff, arow1 + k0 + aoff + 8);

        v16bf b0 = load_frag(brow0 + k0 + boff, brow0 + k0 + boff + 8);
        v16bf b1 = load_frag(brow1 + k0 + boff, brow1 + k0 + boff + 8);
        v16bf b2 = load_frag(brow2 + k0 + boff, brow2 + k0 + boff + 8);
        v16bf b3 = load_frag(brow3 + k0 + boff, brow3 + k0 + boff + 8);

        acc00 = __builtin_amdgcn_wmma_f32_16x16x32_bf16(false, a0, false, b0, (short)0, acc00, false, false);
        acc01 = __builtin_amdgcn_wmma_f32_16x16x32_bf16(false, a0, false, b1, (short)0, acc01, false, false);
        acc02 = __builtin_amdgcn_wmma_f32_16x16x32_bf16(false, a0, false, b2, (short)0, acc02, false, false);
        acc03 = __builtin_amdgcn_wmma_f32_16x16x32_bf16(false, a0, false, b3, (short)0, acc03, false, false);
        acc10 = __builtin_amdgcn_wmma_f32_16x16x32_bf16(false, a1, false, b0, (short)0, acc10, false, false);
        acc11 = __builtin_amdgcn_wmma_f32_16x16x32_bf16(false, a1, false, b1, (short)0, acc11, false, false);
        acc12 = __builtin_amdgcn_wmma_f32_16x16x32_bf16(false, a1, false, b2, (short)0, acc12, false, false);
        acc13 = __builtin_amdgcn_wmma_f32_16x16x32_bf16(false, a1, false, b3, (short)0, acc13, false, false);
    }

    // C/D layout: VGPR v holds M = v + 8*hi, N = lane&15
    const int rbase0 = sm * 32 + hi * 8;
    const int rbase1 = rbase0 + 16;
    const int c0 = sn * 64 + mrow;
    const int c1 = c0 + 16;
    const int c2 = c0 + 32;
    const int c3 = c0 + 48;
#pragma unroll
    for (int v = 0; v < 8; v++) {
        float* e0 = eff + (size_t)(rbase0 + v) * EFF_LD;
        float* e1 = eff + (size_t)(rbase1 + v) * EFF_LD;
        if (c0 < NW) { e0[c0] = acc00[v] + 1.0f; e1[c0] = acc10[v] + 1.0f; }
        if (c1 < NW) { e0[c1] = acc01[v] + 1.0f; e1[c1] = acc11[v] + 1.0f; }
        if (c2 < NW) { e0[c2] = acc02[v] + 1.0f; e1[c2] = acc12[v] + 1.0f; }
        if (c3 < NW) { e0[c3] = acc03[v] + 1.0f; e1[c3] = acc13[v] + 1.0f; }
    }
}

// ---------------- known term: per-row log-softmax at target ----------------
__global__ void known_kernel(const float* __restrict__ eff,
                             const long long* __restrict__ target,
                             float* __restrict__ known_sum) {
    __shared__ float sm[256];
    const int b = blockIdx.x;
    const int tid = threadIdx.x;
    const float* row = eff + (size_t)b * EFF_LD;

    float m = NEGF;
    for (int c = tid; c < NW; c += 256) m = fmaxf(m, row[c]);
    sm[tid] = m; __syncthreads();
    for (int s = 128; s > 0; s >>= 1) {
        if (tid < s) sm[tid] = fmaxf(sm[tid], sm[tid + s]);
        __syncthreads();
    }
    const float rowmax = sm[0]; __syncthreads();

    float acc = 0.0f;
    for (int c = tid; c < NW; c += 256) acc += __expf(row[c] - rowmax);
    sm[tid] = acc; __syncthreads();
    for (int s = 128; s > 0; s >>= 1) {
        if (tid < s) sm[tid] += sm[tid + s];
        __syncthreads();
    }
    if (tid == 0) {
        float lse = __logf(sm[0]) + rowmax;
        long long t = target[b];
        atomicAdd(known_sum, lse - row[t]);
    }
}

// ---------------- novel term: masked logsumexp over gathered logits ----------------
// grid: (G, B/256); one (b,g) per thread; eff gathers hit L2 (8 MB << 192 MB).
__global__ void novel_kernel(const float* __restrict__ eff,
                             const long long* __restrict__ target,
                             const unsigned char* __restrict__ labels_relevant,
                             const long long* __restrict__ rel_idx,
                             const long long* __restrict__ rel_len,
                             float* __restrict__ group_sum,
                             float* __restrict__ cnt) {
    __shared__ int   sidx[RMAXC];
    __shared__ float ssum[256];
    __shared__ float scnt[256];

    const int g   = blockIdx.x;
    const int tid = threadIdx.x;
    const int b   = blockIdx.y * 256 + tid;

    if (tid < RMAXC) sidx[tid] = (int)rel_idx[(size_t)g * RMAXC + tid];
    __syncthreads();

    long long lenl = rel_len[g];
    int len = (int)(lenl < 2 ? 2 : (lenl > RMAXC ? RMAXC : lenl));

    const float* row = eff + (size_t)b * EFF_LD;
    float m = NEGF;
    for (int r = 0; r < len; r++) m = fmaxf(m, row[sidx[r]]);
    float s = 0.0f;
    for (int r = 0; r < len; r++) s += __expf(row[sidx[r]] - m);
    float lse = __logf(s) + m;
    float g0  = row[sidx[0]];

    long long t = target[b];
    bool sel = labels_relevant[(size_t)t * GG + g] != 0;

    ssum[tid] = sel ? (lse - g0) : 0.0f;   // = sel * (-ls0)
    scnt[tid] = sel ? 1.0f : 0.0f;
    __syncthreads();
    for (int st = 128; st > 0; st >>= 1) {
        if (tid < st) { ssum[tid] += ssum[tid + st]; scnt[tid] += scnt[tid + st]; }
        __syncthreads();
    }
    if (tid == 0) {
        if (ssum[0] != 0.0f) atomicAdd(&group_sum[g], ssum[0]);
        if (scnt[0] != 0.0f) atomicAdd(&cnt[g], scnt[0]);
    }
}

// ---------------- finalize: scalar output ----------------
__global__ void finalize_kernel(const float* __restrict__ group_sum,
                                const float* __restrict__ cnt,
                                const float* __restrict__ class_weight,
                                const float* __restrict__ known_sum,
                                float* __restrict__ out) {
    __shared__ float s[512];
    const int tid = threadIdx.x;
    float v = 0.0f;
    if (tid < GG) {
        float c = cnt[tid];
        if (c > 0.0f) v = (group_sum[tid] / fmaxf(c, 1.0f)) * class_weight[tid];
    }
    s[tid] = v; __syncthreads();
    for (int st = 256; st > 0; st >>= 1) {
        if (tid < st) s[tid] += s[tid + st];
        __syncthreads();
    }
    if (tid == 0) out[0] = s[0] + known_sum[0] / (float)BB;
}

// ---------------- launch ----------------
extern "C" void kernel_launch(void* const* d_in, const int* in_sizes, int n_in,
                              void* d_out, int out_size, void* d_ws, size_t ws_size,
                              hipStream_t stream) {
    const float*         inp    = (const float*)d_in[0];
    const float*         gs     = (const float*)d_in[1];
    const float*         cw     = (const float*)d_in[2];
    const long long*     target = (const long long*)d_in[3];
    const unsigned char* lrel   = (const unsigned char*)d_in[4];   // jax bool -> 1 byte
    const long long*     relidx = (const long long*)d_in[5];
    const long long*     rellen = (const long long*)d_in[6];
    float* out = (float*)d_out;

    // workspace layout
    unsigned short* a_bf = (unsigned short*)d_ws;                   // 2048*2048 bf16 (8 MB)
    unsigned short* b_bf = a_bf + (size_t)BB * DD;                  // 1024*2048 bf16 (4 MB)
    float*          eff  = (float*)(b_bf + (size_t)NROWS_B * DD);   // 2048*1024 f32 (8 MB)
    float*          acc  = eff + (size_t)BB * EFF_LD;               // G + G + 1 floats
    float* group_sum = acc;
    float* cnt       = acc + GG;
    float* known_sum = acc + 2 * GG;

    // 1) stage bf16 operands
    {
        int nA = BB * DD;
        cvt_bf16_kernel<<<(nA + 255) / 256, 256, 0, stream>>>(inp, a_bf, nA);
        int nB = NROWS_B * DD;   // first 1024 rows of global_states (S=1860 >= 1024)
        cvt_bf16_kernel<<<(nB + 255) / 256, 256, 0, stream>>>(gs, b_bf, nB);
    }

    // 2) zero accumulators
    init_kernel<<<1, 1024, 0, stream>>>(acc, 2 * GG + 1);

    // 3) WMMA GEMM -> eff (32x64 per wave, 8 WMMAs per k-step)
    gemm_wmma_kernel<<<128, 256, 0, stream>>>(a_bf, b_bf, eff);

    // 4) known term
    known_kernel<<<BB, 256, 0, stream>>>(eff, target, known_sum);

    // 5) novel term
    novel_kernel<<<dim3(GG, BB / 256), 256, 0, stream>>>(
        eff, target, lrel, relidx, rellen, group_sum, cnt);

    // 6) finalize scalar
    finalize_kernel<<<1, 512, 0, stream>>>(group_sum, cnt, cw, known_sum, out);
}
